// dynamicGCN_47820165873709
// MI455X (gfx1250) — compile-verified, matrix operand order checked by reading, not run on hardware
//
#include <hip/hip_runtime.h>
#include <hip/hip_bf16.h>
#include <stddef.h>

typedef __attribute__((ext_vector_type(16))) _Float16 v16h;
typedef __attribute__((ext_vector_type(8)))  _Float16 v8h;
typedef __attribute__((ext_vector_type(8)))  float    v8f;

#define GCN_B 16
#define GCN_N 2048
#define GCN_F 64
#define GCN_H 128

// ---------------------------------------------------------------------------
// Kernel 1: degree + symmetric normalization scale.
// dinv[b,i] = rsqrt(1 + sum_j adj[b,i,j])
// One wave32 per row; float4 loads; shuffle reduction.
// ---------------------------------------------------------------------------
__global__ void gcn_degree_kernel(const float* __restrict__ adj,
                                  float* __restrict__ dinv) {
    const int wave = threadIdx.x >> 5;
    const int lane = threadIdx.x & 31;
    const size_t row = (size_t)blockIdx.x * 8 + wave;   // row in [0, B*N)
    const float4* p = (const float4*)(adj + row * (size_t)GCN_N);
    float s = 0.0f;
#pragma unroll
    for (int i = 0; i < GCN_N / 4 / 32; ++i) {          // 16 float4 per lane
        float4 v = p[lane + i * 32];
        s += v.x + v.y + v.z + v.w;
    }
#pragma unroll
    for (int off = 16; off > 0; off >>= 1) s += __shfl_xor(s, off, 32);
    if (lane == 0) dinv[row] = rsqrtf(s + 1.0f);
}

// ---------------------------------------------------------------------------
// Kernel 2: XWT[b][h][n] = f16( dinv[b][n] * (x @ W1)[b][n][h] )
// Column-scale of the normalized adjacency is folded in here; transposed
// layout gives the aggregation kernel contiguous 16B K-runs.
// ---------------------------------------------------------------------------
__global__ void gcn_feat1_kernel(const float* __restrict__ x,
                                 const float* __restrict__ W1,
                                 const float* __restrict__ dinv,
                                 _Float16* __restrict__ XWT) {
    const int b = blockIdx.y, n = blockIdx.x, h = threadIdx.x; // 128 threads
    __shared__ float xs[GCN_F];
    if (h < GCN_F) xs[h] = x[((size_t)b * GCN_N + n) * GCN_F + h];
    __syncthreads();
    float acc = 0.0f;
#pragma unroll
    for (int f = 0; f < GCN_F; ++f) acc = fmaf(xs[f], W1[f * GCN_H + h], acc);
    XWT[((size_t)b * GCN_H + h) * GCN_N + n] =
        (_Float16)(acc * dinv[(size_t)b * GCN_N + n]);
}

// ---------------------------------------------------------------------------
// Kernel 3: ZT[b][h][n] = f16( dinv[b][n] * (h1 @ W2)[b][n][h] ),  h1 f16 [B,N,H]
// ---------------------------------------------------------------------------
__global__ void gcn_feat2_kernel(const _Float16* __restrict__ h1,
                                 const float* __restrict__ W2,
                                 const float* __restrict__ dinv,
                                 _Float16* __restrict__ ZT) {
    const int b = blockIdx.y, n = blockIdx.x, h = threadIdx.x; // 128 threads
    __shared__ float xs[GCN_H];
    xs[h] = (float)h1[((size_t)b * GCN_N + n) * GCN_H + h];
    __syncthreads();
    float acc = 0.0f;
#pragma unroll
    for (int f = 0; f < GCN_H; ++f) acc = fmaf(xs[f], W2[f * GCN_H + h], acc);
    ZT[((size_t)b * GCN_H + h) * GCN_N + n] =
        (_Float16)(acc * dinv[(size_t)b * GCN_N + n]);
}

// ---------------------------------------------------------------------------
// Kernel 4/5: aggregation
//   out[i,h] = relu( dinv[i] * ( sum_j adj[i,j]*B'[j,h] + B'[i,h] ) + bias[h] )
// where B' already carries the column scale dinv[j]. Inner loop is pure:
// raw adj f32 loads -> f16 A fragment, contiguous B' fragments, 8x
// v_wmma_f32_16x16x32_f16 per K-chunk of 32. Row scale + self-loop + bias
// + relu in the epilogue.
//
// Block = 256 threads = 8 waves; wave w owns rows [blk*128 + 16w, +16) and
// ALL 128 columns (8 accumulators). grid = (N/128, B) = (16, 16).
//
// 16-bit A fragment layout (ISA 7.12.2): lanes 0-15 -> M=lane, K {0..7,16..23};
// lanes 16-31 -> M=lane-16, K {8..15,24..31}. B fragment (32x16) mirrored:
// lane%16 = column, same K striping.
// ---------------------------------------------------------------------------
template <bool FINAL>
__global__ void gcn_aggregate_kernel(const float* __restrict__ adj,
                                     const float* __restrict__ dinv,
                                     const _Float16* __restrict__ BT,
                                     const float* __restrict__ bias,
                                     _Float16* __restrict__ out16,
                                     float* __restrict__ out32) {
    constexpr int N = GCN_N, H = GCN_H;
    const int b      = blockIdx.y;
    const int wave   = threadIdx.x >> 5;
    const int lane   = threadIdx.x & 31;
    const int laneLo = lane & 15;
    const int laneHi = lane >> 4;          // 0/1: selects K sub-stripe
    const int m0     = blockIdx.x * 128 + wave * 16;
    const int m      = m0 + laneLo;        // this lane's A-matrix row

    const float* adjRow = adj + ((size_t)b * N + m) * (size_t)N;
    const _Float16* btBase = BT + (size_t)b * H * N;

    v8f acc[8] = {v8f{}, v8f{}, v8f{}, v8f{}, v8f{}, v8f{}, v8f{}, v8f{}};

    for (int k0 = 0; k0 < N; k0 += 32) {
        const int kA = k0 + laneHi * 8;            // first K run of 8
        const int kB = kA + 16;                    // second K run of 8
        float4 a0 = *(const float4*)(adjRow + kA);
        float4 a1 = *(const float4*)(adjRow + kA + 4);
        float4 a2 = *(const float4*)(adjRow + kB);
        float4 a3 = *(const float4*)(adjRow + kB + 4);

        v16h A;
        A[0]  = (_Float16)a0.x;  A[1]  = (_Float16)a0.y;
        A[2]  = (_Float16)a0.z;  A[3]  = (_Float16)a0.w;
        A[4]  = (_Float16)a1.x;  A[5]  = (_Float16)a1.y;
        A[6]  = (_Float16)a1.z;  A[7]  = (_Float16)a1.w;
        A[8]  = (_Float16)a2.x;  A[9]  = (_Float16)a2.y;
        A[10] = (_Float16)a2.z;  A[11] = (_Float16)a2.w;
        A[12] = (_Float16)a3.x;  A[13] = (_Float16)a3.y;
        A[14] = (_Float16)a3.z;  A[15] = (_Float16)a3.w;

#pragma unroll
        for (int t = 0; t < 8; ++t) {
            const _Float16* bt = btBase + (size_t)(t * 16 + laneLo) * N;
            v8h blo = *(const v8h*)(bt + kA);      // 16B contiguous along K
            v8h bhi = *(const v8h*)(bt + kB);
            v16h Bf;
#pragma unroll
            for (int i = 0; i < 8; ++i) { Bf[i] = blo[i]; Bf[8 + i] = bhi[i]; }
            acc[t] = __builtin_amdgcn_wmma_f32_16x16x32_f16(
                false, A, false, Bf, (short)0, acc[t], false, false);
        }
    }

    // Epilogue. C layout: element (m0 + r + 8*laneHi, t*16 + laneLo).
    const float* dvr = dinv + (size_t)b * N + m0 + 8 * laneHi;
    float4 dr0 = *(const float4*)(dvr);
    float4 dr1 = *(const float4*)(dvr + 4);
    float dr[8] = {dr0.x, dr0.y, dr0.z, dr0.w, dr1.x, dr1.y, dr1.z, dr1.w};

#pragma unroll
    for (int t = 0; t < 8; ++t) {
        const int col = t * 16 + laneLo;
        const float bsv = bias[col];
        // self-loop term B'[row, col] for the 8 rows: contiguous along n in BT
        v8h sv = *(const v8h*)(btBase + (size_t)col * N + m0 + 8 * laneHi);
#pragma unroll
        for (int r = 0; r < 8; ++r) {
            const int row = m0 + r + 8 * laneHi;
            float v = dr[r] * (acc[t][r] + (float)sv[r]) + bsv;
            v = v > 0.0f ? v : 0.0f;               // relu
            const size_t idx = ((size_t)b * N + row) * H + col;
            if (FINAL) out32[idx] = v;
            else       out16[idx] = (_Float16)v;
        }
    }
}

// ---------------------------------------------------------------------------
extern "C" void kernel_launch(void* const* d_in, const int* in_sizes, int n_in,
                              void* d_out, int out_size, void* d_ws, size_t ws_size,
                              hipStream_t stream) {
    (void)in_sizes; (void)n_in; (void)out_size; (void)ws_size;
    const float* x   = (const float*)d_in[0];
    const float* adj = (const float*)d_in[1];
    const float* W1  = (const float*)d_in[2];
    const float* b1  = (const float*)d_in[3];
    const float* W2  = (const float*)d_in[4];
    const float* b2  = (const float*)d_in[5];
    float* out = (float*)d_out;

    // Workspace layout
    char* ws = (char*)d_ws;
    float*    dinv = (float*)ws;                                   // 128 KB
    _Float16* XWT  = (_Float16*)(ws + (size_t)GCN_B * GCN_N * 4);  // 8 MB
    _Float16* h1   = (_Float16*)(ws + (size_t)GCN_B * GCN_N * 4
                                    + (size_t)GCN_B * GCN_H * GCN_N * 2); // 8 MB

    // 1) normalization scales
    gcn_degree_kernel<<<dim3(GCN_B * GCN_N / 8), dim3(256), 0, stream>>>(adj, dinv);
    // 2) B'1 = dinv .* (x @ W1) -> XWT (f16, transposed [B,H,N])
    gcn_feat1_kernel<<<dim3(GCN_N, GCN_B), dim3(GCN_H), 0, stream>>>(x, W1, dinv, XWT);
    // 3) h1 = relu(dinv .* (adj @ B'1 + B'1) + b1) -> f16 [B,N,H]
    gcn_aggregate_kernel<false><<<dim3(GCN_N / 128, GCN_B), dim3(256), 0, stream>>>(
        adj, dinv, XWT, b1, h1, nullptr);
    // 4) B'2 = dinv .* (h1 @ W2) -> ZT (reuse XWT scratch)
    gcn_feat2_kernel<<<dim3(GCN_N, GCN_B), dim3(GCN_H), 0, stream>>>(h1, W2, dinv, XWT);
    // 5) out = relu(dinv .* (adj @ B'2 + B'2) + b2) -> f32 [B,N,H]
    gcn_aggregate_kernel<true><<<dim3(GCN_N / 128, GCN_B), dim3(256), 0, stream>>>(
        adj, dinv, XWT, b2, nullptr, out);
}
